// BitNetMNIST_57939108823182
// MI455X (gfx1250) — matrix operand order repaired and checked by reading, not compiled, not run on hardware
//
#include <hip/hip_runtime.h>
#include <math.h>

typedef __attribute__((ext_vector_type(16))) _Float16 v16h;
typedef __attribute__((ext_vector_type(8)))  _Float16 v8h;
typedef __attribute__((ext_vector_type(8)))  float    v8f;
typedef __attribute__((ext_vector_type(4)))  int      v4i;

#define EPS_Q   1e-5f
#define EPS_RMS 1e-6f

// ---------------------------------------------------------------------------
// CDNA5 async global->LDS copy (16 bytes per lane), tracked by ASYNCcnt.
// Builtin signature (probe-confirmed): (v4i AS1*, v4i AS3*, imm off, imm cpol).
// ---------------------------------------------------------------------------
typedef __attribute__((address_space(1))) v4i as1_v4i;
typedef __attribute__((address_space(3))) v4i as3_v4i;

__device__ __forceinline__ void async_copy16(void* lds, const void* g) {
#if __has_builtin(__builtin_amdgcn_global_load_async_to_lds_b128)
  as1_v4i* gp = (as1_v4i*)(unsigned long long)(g);
  as3_v4i* lp = (as3_v4i*)(unsigned int)(unsigned long long)(lds);
  __builtin_amdgcn_global_load_async_to_lds_b128(gp, lp, 0, 0);
#else
  *(int4*)lds = *(const int4*)g;   // synchronous fallback
#endif
}

__device__ __forceinline__ void wait_async0() {
#if __has_builtin(__builtin_amdgcn_s_wait_asynccnt)
  __builtin_amdgcn_s_wait_asynccnt(0);
#else
  asm volatile("s_wait_asynccnt 0x0" ::: "memory");
#endif
}

// ---------------------------------------------------------------------------
// Weight quantization: absmean scale, ternary round, pad to (Rpad, Cpad), f16.
// ---------------------------------------------------------------------------
__global__ __launch_bounds__(1024)
void quant_weight_kernel(const float* __restrict__ W, _Float16* __restrict__ Wq,
                         int R, int C, int Rpad, int Cpad) {
  __shared__ float red[1024];
  const int tid = threadIdx.x;
  const int n = R * C;
  float s = 0.f;
  for (int i = tid; i < n; i += 1024) s += fabsf(W[i]);
  red[tid] = s;
  __syncthreads();
  #pragma unroll
  for (int off = 512; off > 0; off >>= 1) {
    if (tid < off) red[tid] += red[tid + off];
    __syncthreads();
  }
  const float wscale = fmaxf(red[0] / (float)n, EPS_Q);
  const float inv = 1.f / wscale;
  const int np = Rpad * Cpad;
  for (int i = tid; i < np; i += 1024) {
    const int r = i / Cpad, c = i - r * Cpad;
    float q = 0.f;
    if (r < R && c < C) {
      q = rintf(W[r * C + c] * inv);
      q = fminf(fmaxf(q, -1.f), 1.f);
    }
    Wq[i] = (_Float16)q;
  }
}

// ---------------------------------------------------------------------------
// Per-token absmax activation quantization of x: f32[row,K] -> f16[row,Kpad].
// ---------------------------------------------------------------------------
__global__ __launch_bounds__(256)
void quant_x_kernel(const float* __restrict__ X, _Float16* __restrict__ Xq,
                    float* __restrict__ xscale, int K, int Kpad) {
  const int wave = threadIdx.x >> 5;
  const int lane = threadIdx.x & 31;
  const long long row = (long long)blockIdx.x * 8 + wave;
  const float* xr = X + row * K;
  float am = 0.f;
  for (int k = lane; k < K; k += 32) am = fmaxf(am, fabsf(xr[k]));
  #pragma unroll
  for (int off = 16; off > 0; off >>= 1)
    am = fmaxf(am, __shfl_xor(am, off, 32));
  const float sc = fmaxf(am, EPS_Q);
  const float inv = 1.f / sc;
  _Float16* qr = Xq + row * Kpad;
  for (int k = lane; k < Kpad; k += 32)
    qr[k] = (_Float16)(k < K ? xr[k] * inv : 0.f);
  if (lane == 0) xscale[row] = sc;
}

// ---------------------------------------------------------------------------
// Fused RMSNorm + exact GELU + per-token absmax requant. N = 1024 fixed.
// ---------------------------------------------------------------------------
__global__ __launch_bounds__(256)
void norm_gelu_quant_kernel(const float* __restrict__ H, const float* __restrict__ G,
                            _Float16* __restrict__ Hq, float* __restrict__ hscale) {
  __shared__ float red[256];
  const int N = 1024;
  const int tid = threadIdx.x;
  const long long row = blockIdx.x;
  const float* hr = H + row * N;
  float v[4];
  float ss = 0.f;
  #pragma unroll
  for (int i = 0; i < 4; ++i) { v[i] = hr[tid + 256 * i]; ss += v[i] * v[i]; }
  red[tid] = ss;
  __syncthreads();
  #pragma unroll
  for (int off = 128; off > 0; off >>= 1) {
    if (tid < off) red[tid] += red[tid + off];
    __syncthreads();
  }
  const float rinv = rsqrtf(red[0] * (1.f / N) + EPS_RMS);
  float am = 0.f;
  #pragma unroll
  for (int i = 0; i < 4; ++i) {
    float y = v[i] * rinv * G[tid + 256 * i];
    y = 0.5f * y * (1.f + erff(y * 0.70710678118654752f));
    v[i] = y;
    am = fmaxf(am, fabsf(y));
  }
  __syncthreads();
  red[tid] = am;
  __syncthreads();
  #pragma unroll
  for (int off = 128; off > 0; off >>= 1) {
    if (tid < off) red[tid] = fmaxf(red[tid], red[tid + off]);
    __syncthreads();
  }
  const float sc = fmaxf(red[0], EPS_Q);
  const float inv = 1.f / sc;
  _Float16* qr = Hq + row * N;
  #pragma unroll
  for (int i = 0; i < 4; ++i) qr[tid + 256 * i] = (_Float16)(v[i] * inv);
  if (tid == 0) hscale[blockIdx.x] = sc;
}

// ---------------------------------------------------------------------------
// WMMA GEMM, async global->LDS double-buffered staging, BK=64 per stage.
//   C[M,N] = (A[M,K]f16 @ B[N,K]^T f16) * ascale[m]
// Block = 8 waves (WM x WN); block tile BM x BN; 8 (or 2) WMMAs per barrier.
// LDS rows padded to 72 halves (36 dwords) -> conflict-free b128 row fans.
// Fragment layouts per CDNA5 ISA 7.12.2 (wave32).
// K must be a multiple of 64.
// ---------------------------------------------------------------------------
template <int WM, int WN, int NSUB>
__global__ __launch_bounds__(32 * WM * WN)
void gemm_wmma_lds_kernel(const _Float16* __restrict__ A, const _Float16* __restrict__ B,
                          float* __restrict__ C, const float* __restrict__ ascale,
                          int K, int N_real, int ldc) {
  constexpr int BM  = WM * 16;
  constexpr int BN  = WN * NSUB * 16;
  constexpr int BK  = 64;
  constexpr int LDS_ROW = BK + 8;          // 72 halves = 144 B, 16B-aligned rows
  constexpr int ACH = BM * (BK / 8);       // 16B chunks per A stage
  constexpr int BCH = BN * (BK / 8);
  constexpr int AIT = (ACH + 255) / 256;
  constexpr int BIT = (BCH + 255) / 256;

  __shared__ __attribute__((aligned(32))) _Float16 As[2][BM * LDS_ROW];
  __shared__ __attribute__((aligned(32))) _Float16 Bs[2][BN * LDS_ROW];

  const int tid  = threadIdx.x;
  const int wave = tid >> 5, lane = tid & 31;
  const int wm = wave % WM, wn = wave / WM;
  const int hi = lane >> 4, lo = lane & 15;
  const long long mblock = (long long)blockIdx.x * BM;
  const int nblock = blockIdx.y * BN;

  const _Float16* Ab = A + mblock * K;
  const _Float16* Bb = B + (long long)nblock * K;

  auto stage_in = [&](int st, int k0) {
    #pragma unroll
    for (int i = 0; i < AIT; ++i) {
      const int c = tid + i * 256;
      bool go = true;
      if constexpr (ACH % 256 != 0) go = (c < ACH);
      if (go) {
        const int row = c >> 3, col = (c & 7) * 8;
        async_copy16(&As[st][row * LDS_ROW + col], Ab + (long long)row * K + k0 + col);
      }
    }
    #pragma unroll
    for (int i = 0; i < BIT; ++i) {
      const int c = tid + i * 256;
      bool go = true;
      if constexpr (BCH % 256 != 0) go = (c < BCH);
      if (go) {
        const int row = c >> 3, col = (c & 7) * 8;
        async_copy16(&Bs[st][row * LDS_ROW + col], Bb + (long long)row * K + k0 + col);
      }
    }
  };

  v8f acc[NSUB] = {};

  // Prologue: stage 0 resident before first compute.
  stage_in(0, 0);
  wait_async0();
  __syncthreads();

  const int nstages = K / BK;
  int st = 0;
  for (int sidx = 0; sidx < nstages; ++sidx) {
    // Kick off next stage's async copies; they overlap the WMMA work below.
    if (sidx + 1 < nstages) stage_in(st ^ 1, (sidx + 1) * BK);

    #pragma unroll
    for (int kk = 0; kk < BK; kk += 32) {
      // A fragment from LDS.
      const _Float16* As_w = &As[st][(wm * 16 + lo) * LDS_ROW + kk];
      const v8h c0 = *(const v8h*)(As_w + hi * 8);
      const v8h c1 = *(const v8h*)(As_w + 16 + hi * 8);
      const v16h a = __builtin_shufflevector(c0, c1, 0, 1, 2, 3, 4, 5, 6, 7,
                                             8, 9, 10, 11, 12, 13, 14, 15);
      // Batch all B fragments first so ds waits amortize across WMMAs.
      v16h b[NSUB];
      #pragma unroll
      for (int s = 0; s < NSUB; ++s)
      {
        const _Float16* Bs_w = &Bs[st][(wn * NSUB * 16 + s * 16 + lo) * LDS_ROW + kk];
        const v8h d0 = *(const v8h*)(Bs_w + hi * 8);
        const v8h d1 = *(const v8h*)(Bs_w + 16 + hi * 8);
        b[s] = __builtin_shufflevector(d0, d1, 0, 1, 2, 3, 4, 5, 6, 7,
                                       8, 9, 10, 11, 12, 13, 14, 15);
      }
      #pragma unroll
      for (int s = 0; s < NSUB; ++s)
        acc[s] = __builtin_amdgcn_wmma_f32_16x16x32_f16(
            /*neg_a=*/false, a, /*neg_b=*/false, b[s],
            /*c_mod=*/(short)0, acc[s], /*reuse_a=*/false, /*reuse_b=*/false);
    }

    wait_async0();     // next stage landed in LDS
    __syncthreads();   // all waves done reading current stage
    st ^= 1;
  }

  float asc[8];
  #pragma unroll
  for (int v = 0; v < 8; ++v) asc[v] = ascale[mblock + wm * 16 + hi * 8 + v];

  #pragma unroll
  for (int s = 0; s < NSUB; ++s) {
    const int col = nblock + wn * NSUB * 16 + s * 16 + lo;
    if (col < N_real) {
      #pragma unroll
      for (int v = 0; v < 8; ++v) {
        const long long row = mblock + wm * 16 + hi * 8 + v;
        C[row * ldc + col] = acc[s][v] * asc[v];
      }
    }
  }
}

// ---------------------------------------------------------------------------
extern "C" void kernel_launch(void* const* d_in, const int* in_sizes, int n_in,
                              void* d_out, int out_size, void* d_ws, size_t ws_size,
                              hipStream_t stream) {
  const float* x  = (const float*)d_in[0];
  const float* w1 = (const float*)d_in[1];
  const float* g1 = (const float*)d_in[2];
  const float* w2 = (const float*)d_in[3];
  const float* g2 = (const float*)d_in[4];
  const float* w3 = (const float*)d_in[5];
  float* out = (float*)d_out;

  const int B = 65536, IN = 784, INP = 832, H = 1024, ON = 10;  // INP = 13*64

  char* p = (char*)d_ws;
  auto alloc = [&](size_t bytes) -> char* {
    char* r = p;
    p += (bytes + 255) & ~(size_t)255;
    return r;
  };
  _Float16* w1q = (_Float16*)alloc((size_t)H * INP * 2);
  _Float16* w2q = (_Float16*)alloc((size_t)H * H * 2);
  _Float16* w3q = (_Float16*)alloc((size_t)16 * H * 2);
  _Float16* xq  = (_Float16*)alloc((size_t)B * INP * 2);
  float*    xsc = (float*)   alloc((size_t)B * 4);
  float*    h   = (float*)   alloc((size_t)B * H * 4);
  _Float16* hq  = (_Float16*)alloc((size_t)B * H * 2);
  float*    hsc = (float*)   alloc((size_t)B * 4);

  // Ternary weight quantization (quantized weights stay L2-resident: <4 MB).
  quant_weight_kernel<<<1, 1024, 0, stream>>>(w1, w1q, H, IN, H, INP);
  quant_weight_kernel<<<1, 1024, 0, stream>>>(w2, w2q, H, H, H, H);
  quant_weight_kernel<<<1, 1024, 0, stream>>>(w3, w3q, ON, H, 16, H);

  // Per-token activation quantization of x (pad K 784 -> 832).
  quant_x_kernel<<<B / 8, 256, 0, stream>>>(x, xq, xsc, IN, INP);

  // Layer 1: h = (xq @ w1q^T) * xsc   [65536 x 1024], K = 832
  gemm_wmma_lds_kernel<4, 2, 4><<<dim3(B / 64, H / 128), 256, 0, stream>>>(
      xq, w1q, h, xsc, INP, H, H);
  norm_gelu_quant_kernel<<<B, 256, 0, stream>>>(h, g1, hq, hsc);

  // Layer 2: h = (hq @ w2q^T) * hsc   [65536 x 1024], K = 1024
  gemm_wmma_lds_kernel<4, 2, 4><<<dim3(B / 64, H / 128), 256, 0, stream>>>(
      hq, w2q, h, hsc, H, H, H);
  norm_gelu_quant_kernel<<<B, 256, 0, stream>>>(h, g2, hq, hsc);

  // Layer 3: out = (hq @ w3q^T) * hsc  [65536 x 10], N padded to 16
  gemm_wmma_lds_kernel<8, 1, 1><<<dim3(B / 128, 1), 256, 0, stream>>>(
      hq, w3q, out, hsc, H, ON, ON);
}
